// Qwen2MoELayer_80676665688478
// MI455X (gfx1250) — compile-verified, matrix operand order checked by reading, not run on hardware
//
#include <hip/hip_runtime.h>
#include <hip/hip_bf16.h>
#include <cstdint>

typedef __bf16 bf16;
typedef bf16  v16bf __attribute__((ext_vector_type(16)));
typedef bf16  v8bf  __attribute__((ext_vector_type(8)));
typedef float v8f   __attribute__((ext_vector_type(8)));

#define T_TOK 8192
#define HID   2048
#define INTER 1408
#define NEXP  8

#define BM 128
#define BN 64
#define BK 64
#define LDH 72   // halfword row stride in LDS (64 data + 8 pad -> 16B-aligned chunks, conflict-free)

static __device__ __forceinline__ uint32_t pack2bf(float a, float b) {
  uint32_t ua = __builtin_bit_cast(uint32_t, a);
  uint32_t ub = __builtin_bit_cast(uint32_t, b);
  return (ub & 0xFFFF0000u) | (ua >> 16);
}
static __device__ __forceinline__ unsigned short f2bf(float a) {
  return (unsigned short)(__builtin_bit_cast(uint32_t, a) >> 16);
}
// Stitch two contiguous 16B LDS chunks into one 16x32 bf16 fragment register set.
static __device__ __forceinline__ v16bf frag16(const unsigned short* p0) {
  v8bf lo = *(const v8bf*)p0;
  v8bf hi = *(const v8bf*)(p0 + 16);
  return __builtin_shufflevector(lo, hi, 0,1,2,3,4,5,6,7,8,9,10,11,12,13,14,15);
}

// ---------------- Router: logits -> softmax -> top2 -> dense gates [T, E] ----------------
__global__ __launch_bounds__(256)
void router_kernel(const float* __restrict__ x,
                   const float* __restrict__ rw,
                   float* __restrict__ gates) {
  const int wave = threadIdx.x >> 5;
  const int lane = threadIdx.x & 31;
  const int t = blockIdx.x * 8 + wave;

  float acc[NEXP];
#pragma unroll
  for (int e = 0; e < NEXP; ++e) acc[e] = 0.f;

  const float* xr = x + (size_t)t * HID;
  for (int k = lane; k < HID; k += 32) {
    float xv = xr[k];
#pragma unroll
    for (int e = 0; e < NEXP; ++e) acc[e] += xv * rw[e * HID + k];
  }
#pragma unroll
  for (int e = 0; e < NEXP; ++e) {
#pragma unroll
    for (int off = 16; off > 0; off >>= 1)
      acc[e] += __shfl_xor(acc[e], off, 32);
  }
  if (lane == 0) {
    float mx = acc[0];
#pragma unroll
    for (int e = 1; e < NEXP; ++e) mx = fmaxf(mx, acc[e]);
    float p[NEXP];
    float s = 0.f;
#pragma unroll
    for (int e = 0; e < NEXP; ++e) { p[e] = __expf(acc[e] - mx); s += p[e]; }
    const float inv = 1.f / s;
#pragma unroll
    for (int e = 0; e < NEXP; ++e) p[e] *= inv;
    int i0 = 0;
#pragma unroll
    for (int e = 1; e < NEXP; ++e) if (p[e] > p[i0]) i0 = e;
    int i1 = (i0 == 0) ? 1 : 0;
#pragma unroll
    for (int e = 0; e < NEXP; ++e) if (e != i0 && p[e] > p[i1]) i1 = e;
#pragma unroll
    for (int e = 0; e < NEXP; ++e)
      gates[(size_t)t * NEXP + e] = (e == i0 || e == i1) ? p[e] : 0.f;
  }
}

// ---------------- GEMM1: gu = x @ Wgu[e], SwiGLU epilogue -> bf16 h1 [T, I] ----------------
// Ping-pong double-buffered LDS; next-tile global loads issued before the WMMA block.
__global__ __launch_bounds__(256)
void gateup_swiglu_kernel(const float* __restrict__ x,
                          const float* __restrict__ wgu,      // [HID][2*INTER] (this expert)
                          unsigned short* __restrict__ h1) {  // bf16 [T][INTER]
  __shared__ unsigned short sA[2][BM * LDH];
  __shared__ unsigned short sB[2][2][BN * LDH];

  const int tid  = threadIdx.x;
  const int lane = tid & 31;
  const int wave = tid >> 5;
  const int wm = (wave & 3) * 32;   // 4 waves along M
  const int wn = (wave >> 2) * 32;  // 2 waves along N
  const int m0 = blockIdx.y * BM;
  const int n0 = blockIdx.x * BN;

  v8f cg[2][2], cu[2][2];
#pragma unroll
  for (int i = 0; i < 2; ++i)
#pragma unroll
    for (int j = 0; j < 2; ++j) { cg[i][j] = v8f{}; cu[i][j] = v8f{}; }

  float4 ra[8];          // A staging: 128x64 fp32 = 8 float4/thread
  float4 rbg[4], rbu[4]; // B staging: 64x64 fp32 per tile = 4 float4/thread each

  auto load_regs = [&](int kk) {
#pragma unroll
    for (int i = 0; i < 8; ++i) {
      int f = tid + i * 256, row = f >> 4, c4 = f & 15;
      ra[i] = *(const float4*)(x + (size_t)(m0 + row) * HID + kk + c4 * 4);
    }
#pragma unroll
    for (int i = 0; i < 4; ++i) {
      int f = tid + i * 256, k = f >> 4, c4 = f & 15;
      const float* srow = wgu + (size_t)(kk + k) * (2 * INTER) + n0 + c4 * 4;
      rbg[i] = *(const float4*)(srow);
      rbu[i] = *(const float4*)(srow + INTER);
    }
  };
  auto store_lds = [&](int b) {
#pragma unroll
    for (int i = 0; i < 8; ++i) {
      int f = tid + i * 256, row = f >> 4, c4 = f & 15;
      uint32_t* dst = (uint32_t*)&sA[b][row * LDH + c4 * 4];
      dst[0] = pack2bf(ra[i].x, ra[i].y);
      dst[1] = pack2bf(ra[i].z, ra[i].w);
    }
#pragma unroll
    for (int i = 0; i < 4; ++i) {
      int f = tid + i * 256, k = f >> 4, c4 = f & 15;
      float g4[4] = {rbg[i].x, rbg[i].y, rbg[i].z, rbg[i].w};
      float u4[4] = {rbu[i].x, rbu[i].y, rbu[i].z, rbu[i].w};
#pragma unroll
      for (int j = 0; j < 4; ++j) {
        sB[b][0][(c4 * 4 + j) * LDH + k] = f2bf(g4[j]);
        sB[b][1][(c4 * 4 + j) * LDH + k] = f2bf(u4[j]);
      }
    }
  };
  auto compute = [&](int b) {
#pragma unroll
    for (int ks = 0; ks < BK; ks += 32) {
      const int khalf = (lane >> 4) * 8;
      const int r16   = lane & 15;
      v16bf a[2], bg[2], bu[2];
#pragma unroll
      for (int mi = 0; mi < 2; ++mi)
        a[mi] = frag16(&sA[b][(wm + mi * 16 + r16) * LDH + ks + khalf]);
#pragma unroll
      for (int ni = 0; ni < 2; ++ni) {
        bg[ni] = frag16(&sB[b][0][(wn + ni * 16 + r16) * LDH + ks + khalf]);
        bu[ni] = frag16(&sB[b][1][(wn + ni * 16 + r16) * LDH + ks + khalf]);
      }
#pragma unroll
      for (int mi = 0; mi < 2; ++mi)
#pragma unroll
        for (int ni = 0; ni < 2; ++ni) {
          cg[mi][ni] = __builtin_amdgcn_wmma_f32_16x16x32_bf16(
              false, a[mi], false, bg[ni], (short)0, cg[mi][ni], false, false);
          cu[mi][ni] = __builtin_amdgcn_wmma_f32_16x16x32_bf16(
              false, a[mi], false, bu[ni], (short)0, cu[mi][ni], false, false);
        }
    }
  };

  constexpr int NK = HID / BK;
  load_regs(0);
  store_lds(0);
  __syncthreads();
  int cb = 0;
  for (int kt = 0; kt < NK; ++kt) {
    const bool more = (kt + 1) < NK;
    if (more) load_regs((kt + 1) * BK);   // global loads overlap the WMMAs below
    compute(cb);
    if (more) store_lds(cb ^ 1);          // cvt + write the other buffer
    __syncthreads();
    cb ^= 1;
  }

  // Epilogue: silu(g) * u -> bf16
  const int mAdd = (lane < 16) ? 0 : 8;
  const int nIn  = lane & 15;
#pragma unroll
  for (int mi = 0; mi < 2; ++mi)
#pragma unroll
    for (int ni = 0; ni < 2; ++ni)
#pragma unroll
      for (int r = 0; r < 8; ++r) {
        const int gm = m0 + wm + mi * 16 + r + mAdd;
        const int gn = n0 + wn + ni * 16 + nIn;
        const float g = cg[mi][ni][r];
        const float u = cu[mi][ni][r];
        const float h = (g / (1.f + __expf(-g))) * u;
        h1[(size_t)gm * INTER + gn] = f2bf(h);
      }
}

// ---------------- GEMM2: out (+)= gates[:,e] * (h1 @ Wd[e]) ----------------
// A tile (bf16, pure copy) moved with GLOBAL_LOAD_ASYNC_TO_LDS_B128 (ASYNCcnt path).
__global__ __launch_bounds__(256)
void down_combine_kernel(const unsigned short* __restrict__ h1,  // bf16 [T][INTER]
                         const float* __restrict__ wd,           // [INTER][HID] (this expert)
                         const float* __restrict__ gates,        // [T][NEXP]
                         float* __restrict__ out,
                         int e, int first) {
  __shared__ unsigned short sA[2][BM * LDH];
  __shared__ unsigned short sB[2][BN * LDH];

  const int tid  = threadIdx.x;
  const int lane = tid & 31;
  const int wave = tid >> 5;
  const int wm = (wave & 3) * 32;
  const int wn = (wave >> 2) * 32;
  const int m0 = blockIdx.y * BM;
  const int n0 = blockIdx.x * BN;

  v8f c[2][2];
#pragma unroll
  for (int i = 0; i < 2; ++i)
#pragma unroll
    for (int j = 0; j < 2; ++j) c[i][j] = v8f{};

  float4 rb[4];

  auto issue_asyncA = [&](int b, int kk) {
#pragma unroll
    for (int i = 0; i < 4; ++i) {
      int f = tid + i * 256, row = f >> 3, c8 = f & 7;
      const unsigned short* g = h1 + (size_t)(m0 + row) * INTER + kk + c8 * 8;
      uint32_t l = (uint32_t)(uintptr_t)&sA[b][row * LDH + c8 * 8];
      asm volatile("global_load_async_to_lds_b128 %0, %1, off"
                   :: "v"(l), "v"((unsigned long long)(uintptr_t)g)
                   : "memory");
    }
  };
  auto load_regsB = [&](int kk) {
#pragma unroll
    for (int i = 0; i < 4; ++i) {
      int f = tid + i * 256, k = f >> 4, c4 = f & 15;
      rb[i] = *(const float4*)(wd + (size_t)(kk + k) * HID + n0 + c4 * 4);
    }
  };
  auto store_ldsB = [&](int b) {
#pragma unroll
    for (int i = 0; i < 4; ++i) {
      int f = tid + i * 256, k = f >> 4, c4 = f & 15;
      float q[4] = {rb[i].x, rb[i].y, rb[i].z, rb[i].w};
#pragma unroll
      for (int j = 0; j < 4; ++j)
        sB[b][(c4 * 4 + j) * LDH + k] = f2bf(q[j]);
    }
  };
  auto compute = [&](int b) {
#pragma unroll
    for (int ks = 0; ks < BK; ks += 32) {
      const int khalf = (lane >> 4) * 8;
      const int r16   = lane & 15;
      v16bf a[2], bb[2];
#pragma unroll
      for (int mi = 0; mi < 2; ++mi)
        a[mi] = frag16(&sA[b][(wm + mi * 16 + r16) * LDH + ks + khalf]);
#pragma unroll
      for (int ni = 0; ni < 2; ++ni)
        bb[ni] = frag16(&sB[b][(wn + ni * 16 + r16) * LDH + ks + khalf]);
#pragma unroll
      for (int mi = 0; mi < 2; ++mi)
#pragma unroll
        for (int ni = 0; ni < 2; ++ni)
          c[mi][ni] = __builtin_amdgcn_wmma_f32_16x16x32_bf16(
              false, a[mi], false, bb[ni], (short)0, c[mi][ni], false, false);
    }
  };

  constexpr int NK = INTER / BK;
  issue_asyncA(0, 0);
  load_regsB(0);
  store_ldsB(0);
  asm volatile("s_wait_asynccnt 0x0" ::: "memory");
  __syncthreads();
  int cb = 0;
  for (int kt = 0; kt < NK; ++kt) {
    const bool more = (kt + 1) < NK;
    if (more) {
      issue_asyncA(cb ^ 1, (kt + 1) * BK);  // async DMA overlaps the WMMAs
      load_regsB((kt + 1) * BK);
    }
    compute(cb);
    if (more) store_ldsB(cb ^ 1);
    asm volatile("s_wait_asynccnt 0x0" ::: "memory");
    __syncthreads();
    cb ^= 1;
  }

  // Epilogue: scale by gate, accumulate into out (beta = first ? 0 : 1)
  const int mAdd = (lane < 16) ? 0 : 8;
  const int nIn  = lane & 15;
#pragma unroll
  for (int mi = 0; mi < 2; ++mi)
#pragma unroll
    for (int ni = 0; ni < 2; ++ni)
#pragma unroll
      for (int r = 0; r < 8; ++r) {
        const int gm = m0 + wm + mi * 16 + r + mAdd;
        const int gn = n0 + wn + ni * 16 + nIn;
        const float gate = gates[(size_t)gm * NEXP + e];
        const float val  = gate * c[mi][ni][r];
        const size_t oi  = (size_t)gm * HID + gn;
        if (first) out[oi] = val;
        else       out[oi] += val;
      }
}

extern "C" void kernel_launch(void* const* d_in, const int* in_sizes, int n_in,
                              void* d_out, int out_size, void* d_ws, size_t ws_size,
                              hipStream_t stream) {
  const float* x   = (const float*)d_in[0];
  const float* rw  = (const float*)d_in[1];
  const float* wgu = (const float*)d_in[2];
  const float* wd  = (const float*)d_in[3];
  float* out = (float*)d_out;

  float* gates = (float*)d_ws;
  size_t gbytes = ((size_t)T_TOK * NEXP * sizeof(float) + 255) & ~(size_t)255;
  unsigned short* h1 = (unsigned short*)((char*)d_ws + gbytes);  // bf16 [T][INTER]

  // 1. Router gates
  router_kernel<<<T_TOK / 8, 256, 0, stream>>>(x, rw, gates);

  // 2-3. Dense per-expert MLP with masked (gated) combine
  for (int e = 0; e < NEXP; ++e) {
    const float* wgu_e = wgu + (size_t)e * HID * (2 * INTER);
    const float* wd_e  = wd  + (size_t)e * INTER * HID;

    dim3 g1(INTER / BN, T_TOK / BM);   // 22 x 64
    gateup_swiglu_kernel<<<g1, 256, 0, stream>>>(x, wgu_e, h1);

    dim3 g2(HID / BN, T_TOK / BM);     // 32 x 64
    down_combine_kernel<<<g2, 256, 0, stream>>>(h1, wd_e, gates, out, e, e == 0 ? 1 : 0);
  }
}
